// DIEN_24094766530912
// MI455X (gfx1250) — compile-verified, matrix operand order checked by reading, not run on hardware
//
#include <hip/hip_runtime.h>

// ---------------------------------------------------------------------------
// DIEN inference for MI455X (gfx1250, wave32, WMMA bf16 16x16x32)
// ---------------------------------------------------------------------------

#define BB   2048
#define TT   200
#define HH   128
#define G3   384   // 3*H
#define JOIN 320

typedef __attribute__((ext_vector_type(16))) __bf16 bf16x16;
typedef __attribute__((ext_vector_type(8)))  float  f32x8;

union TileBF {
  bf16x16 v;
  uint4   q[2];
  unsigned short s[16];
};

static __device__ __forceinline__ unsigned short f32_to_bf16(float f) {
  unsigned int u = __float_as_uint(f);
  u = (u + 0x7FFFu + ((u >> 16) & 1u)) >> 16;
  return (unsigned short)u;
}
static __device__ __forceinline__ float bf16_to_f32(unsigned short u) {
  return __uint_as_float(((unsigned int)u) << 16);
}
static __device__ __forceinline__ float sigmoidf(float x) {
  return 1.0f / (1.0f + __expf(-x));
}

// Swizzled storage for a K x N (or K x M for A-operands) 16-bit matrix so that
// each lane of a wave can fetch its 16 WMMA operand elements contiguously.
// ISA 16-bit operand layout (16x32 A / 32x16 B): lane = {khalf,col}, per-lane
// elements e0..7 = K(khalf*8 .. +7), e8..15 = K(16+khalf*8 .. +7).
static __device__ __forceinline__ int swz_index(int k, int col, int K) {
  int ct = col >> 4, cc = col & 15;
  int kt = k >> 5,  kk = k & 31;
  int lhalf = (kk >> 3) & 1;
  int e     = (kk & 7) + ((kk >> 4) << 3);
  int lane  = lhalf * 16 + cc;
  int tile  = ct * (K >> 5) + kt;
  return tile * 512 + lane * 16 + e;
}

static __device__ __forceinline__ bf16x16 load_tile_swz(const unsigned short* tile, int lane) {
  TileBF t;
  const unsigned short* p = tile + lane * 16;
  t.q[0] = *(const uint4*)(p);
  t.q[1] = *(const uint4*)(p + 8);
  return t.v;
}

// A-tile (16x32) directly from a row-major bf16 matrix in global memory.
// basePtr points at element (row0, k0); rowStride in elements.
static __device__ __forceinline__ bf16x16 load_tileA_global(const unsigned short* basePtr,
                                                            size_t rowStride, int lane) {
  int m = lane & 15;
  int khalf = (lane >> 4) << 3;      // 0 or 8
  const unsigned short* r = basePtr + (size_t)m * rowStride + khalf;
  TileBF t;
  t.q[0] = *(const uint4*)(r);
  t.q[1] = *(const uint4*)(r + 16);
  return t.v;
}

static __device__ __forceinline__ f32x8 wmma_bf16(bf16x16 a, bf16x16 b, f32x8 c) {
  return __builtin_amdgcn_wmma_f32_16x16x32_bf16(false, a, false, b, (short)0, c, false, false);
}

// ---------------------------------------------------------------------------
// K0: f32 -> bf16 weight conversion
// ---------------------------------------------------------------------------
__global__ void cvt_bf16_kernel(const float* __restrict__ src,
                                unsigned short* __restrict__ dst, int n) {
  int i = blockIdx.x * 256 + threadIdx.x;
  if (i < n) dst[i] = f32_to_bf16(src[i]);
}

// ---------------------------------------------------------------------------
// K1a: hist_join gather: [B,T,128] bf16 = [item_emb | cate_emb[cate_list]]
// ---------------------------------------------------------------------------
__global__ __launch_bounds__(128) void gather_hist_kernel(
    const int* __restrict__ history, const int* __restrict__ cate_list,
    const float* __restrict__ item_emb, const float* __restrict__ cate_emb,
    unsigned short* __restrict__ histj) {
  int row = blockIdx.x;          // b*T + t
  int h   = threadIdx.x;         // 0..127
  int it  = history[row];
  float v = (h < 64) ? item_emb[(size_t)it * 64 + h]
                     : cate_emb[(size_t)cate_list[it] * 64 + (h - 64)];
  histj[(size_t)row * HH + h] = f32_to_bf16(v);
}

// K1b: item_join gather: [B,128] f32
__global__ __launch_bounds__(128) void gather_item_kernel(
    const int* __restrict__ item, const int* __restrict__ cate_list,
    const float* __restrict__ item_emb, const float* __restrict__ cate_emb,
    float* __restrict__ item_join) {
  int b = blockIdx.x, h = threadIdx.x;
  int it = item[b];
  float v = (h < 64) ? item_emb[(size_t)it * 64 + h]
                     : cate_emb[(size_t)cate_list[it] * 64 + (h - 64)];
  item_join[(size_t)b * HH + h] = v;
}

// ---------------------------------------------------------------------------
// K2: fused GRU (reset_after) over T steps. One block owns 16 batch rows and
// keeps both 128x384 weight matrices swizzled in LDS (CDNA5: 320KB/WGP).
// ---------------------------------------------------------------------------
__global__ __launch_bounds__(256) void gru_kernel(
    const unsigned short* __restrict__ histj,   // [B,T,H] bf16
    const unsigned short* __restrict__ wx_g,    // [H,3H] bf16 row-major
    const unsigned short* __restrict__ wh_g,    // [H,3H] bf16 row-major
    const float* __restrict__ b_in,
    const float* __restrict__ b_rec,
    unsigned short* __restrict__ gseq) {        // [T,B,H] bf16
  extern __shared__ unsigned char smem[];
  unsigned short* wx  = (unsigned short*)smem;          // 49152 shorts (swizzled)
  unsigned short* wh  = wx + HH * G3;                   // 49152 shorts
  unsigned short* hbf = wh + HH * G3;                   // 2048 shorts (A-swizzled h)
  float* hst = (float*)(hbf + 16 * HH);                 // 2048 f32 (h state)
  float* gx  = hst + 16 * HH;                           // 6144 f32 (x @ Wx)
  float* gh  = gx + 16 * G3;                            // 6144 f32 (h @ Wh)

  const int tid = threadIdx.x;
  const int b0  = blockIdx.x * 16;
  const int wave = tid >> 5, lane = tid & 31;
  const int mrow = lane & 15;
  const int khalf = (lane >> 4) << 3;
  const int rbase = (lane >> 4) * 8;

  for (int i = tid; i < HH * G3; i += 256) {
    int k = i / G3, n = i % G3;
    int si = swz_index(k, n, HH);
    wx[si] = wx_g[i];
    wh[si] = wh_g[i];
  }
  for (int i = tid; i < 16 * HH; i += 256) { hbf[i] = 0; hst[i] = 0.0f; }
  __syncthreads();

  for (int t = 0; t < TT; ++t) {
    // ---- hoisted A operands: x_t (global) and h (LDS), K = 0..127 ----
    bf16x16 ax[4], ah[4];
    const unsigned short* xbase = histj + ((size_t)b0 * TT + t) * HH;
    for (int kt = 0; kt < 4; ++kt) {
      const unsigned short* r = xbase + (size_t)mrow * ((size_t)TT * HH) + kt * 32 + khalf;
      TileBF tx;
      tx.q[0] = *(const uint4*)(r);
      tx.q[1] = *(const uint4*)(r + 16);
      ax[kt] = tx.v;
      ah[kt] = load_tile_swz(hbf + kt * 512, lane);
    }
    // ---- 24 N-tiles of [16,384]; 3 per wave; dual accumulators ----
    for (int s = 0; s < 3; ++s) {
      int nt = wave * 3 + s;
      f32x8 accx = {}; f32x8 acch = {};
      for (int kt = 0; kt < 4; ++kt) {
        bf16x16 bx = load_tile_swz(wx + (nt * 4 + kt) * 512, lane);
        bf16x16 bh = load_tile_swz(wh + (nt * 4 + kt) * 512, lane);
        accx = wmma_bf16(ax[kt], bx, accx);
        acch = wmma_bf16(ah[kt], bh, acch);
      }
      int col = nt * 16 + (lane & 15);
      for (int i = 0; i < 8; ++i) {
        gx[(rbase + i) * G3 + col] = accx[i];
        gh[(rbase + i) * G3 + col] = acch[i];
      }
    }
    __syncthreads();
    // ---- gates + state update ----
    for (int e = tid; e < 16 * HH; e += 256) {
      int m = e >> 7, h = e & 127;
      float xz = gx[m * G3 + h]        + b_in[h];
      float hz = gh[m * G3 + h]        + b_rec[h];
      float xr = gx[m * G3 + HH + h]   + b_in[HH + h];
      float hr = gh[m * G3 + HH + h]   + b_rec[HH + h];
      float xh = gx[m * G3 + 2*HH + h] + b_in[2*HH + h];
      float hc = gh[m * G3 + 2*HH + h] + b_rec[2*HH + h];
      float z = sigmoidf(xz + hz);
      float r = sigmoidf(xr + hr);
      float cand = tanhf(xh + r * hc);
      float hn = z * hst[e] + (1.0f - z) * cand;
      hst[e] = hn;
      unsigned short hb = f32_to_bf16(hn);
      hbf[swz_index(h, m, HH)] = hb;
      gseq[((size_t)t * BB + (b0 + m)) * HH + h] = hb;
    }
    __syncthreads();
  }
}

// ---------------------------------------------------------------------------
// K3: attention scores + softmax over T (no mask, matching reference)
// ---------------------------------------------------------------------------
__global__ __launch_bounds__(256) void attn_kernel(
    const unsigned short* __restrict__ gseq,    // [T,B,H] bf16
    const float* __restrict__ item_join,        // [B,H] f32
    float* __restrict__ attn) {                 // [B,T]
  __shared__ float itj[HH];
  __shared__ float sc[256];
  __shared__ float red[256];
  int b = blockIdx.x, tid = threadIdx.x;
  if (tid < HH) itj[tid] = item_join[(size_t)b * HH + tid];
  __syncthreads();
  float val = 0.0f, mval = -1e30f;
  if (tid < TT) {
    const unsigned short* row = gseq + ((size_t)tid * BB + b) * HH;
    float acc = 0.0f;
    for (int h = 0; h < HH; ++h) acc += itj[h] * bf16_to_f32(row[h]);
    val = acc; mval = acc;
  }
  sc[tid] = val; red[tid] = mval;
  __syncthreads();
  for (int off = 128; off > 0; off >>= 1) {
    if (tid < off) red[tid] = fmaxf(red[tid], red[tid + off]);
    __syncthreads();
  }
  float mx = red[0];
  __syncthreads();
  float ex = (tid < TT) ? __expf(sc[tid] - mx) : 0.0f;
  red[tid] = ex;
  __syncthreads();
  for (int off = 128; off > 0; off >>= 1) {
    if (tid < off) red[tid] += red[tid + off];
    __syncthreads();
  }
  float inv = 1.0f / red[0];
  if (tid < TT) attn[(size_t)b * TT + tid] = ex * inv;
}

// ---------------------------------------------------------------------------
// K4: fused AUGRU over T steps; 5x (128x128) weights swizzled in LDS.
// ---------------------------------------------------------------------------
__global__ __launch_bounds__(256) void augru_kernel(
    const unsigned short* __restrict__ gseq,    // [T,B,H] bf16
    const float* __restrict__ attn,             // [B,T]
    const unsigned short* __restrict__ wu_g, const unsigned short* __restrict__ uu_g,
    const unsigned short* __restrict__ wr_g, const unsigned short* __restrict__ ur_g,
    const unsigned short* __restrict__ uc_g,
    const float* __restrict__ bu, const float* __restrict__ br,
    float* __restrict__ hist_hid) {             // [B,H]
  extern __shared__ unsigned char smem[];
  unsigned short* wu = (unsigned short*)smem;   // each 16384 shorts (swizzled)
  unsigned short* uu = wu + HH * HH;
  unsigned short* wr = uu + HH * HH;
  unsigned short* ur = wr + HH * HH;
  unsigned short* uc = ur + HH * HH;
  unsigned short* hbf = uc + HH * HH;           // 2048 shorts
  float* hst = (float*)(hbf + 16 * HH);         // 2048 f32
  float* gates = hst + 16 * HH;                 // 6144 f32: [u | r | hc]

  const int tid = threadIdx.x;
  const int b0  = blockIdx.x * 16;
  const int wave = tid >> 5, lane = tid & 31;
  const int mrow = lane & 15;
  const int khalf = (lane >> 4) << 3;
  const int rbase = (lane >> 4) * 8;

  for (int i = tid; i < HH * HH; i += 256) {
    int k = i / HH, n = i % HH;
    int si = swz_index(k, n, HH);
    wu[si] = wu_g[i]; uu[si] = uu_g[i];
    wr[si] = wr_g[i]; ur[si] = ur_g[i];
    uc[si] = uc_g[i];
  }
  for (int i = tid; i < 16 * HH; i += 256) { hbf[i] = 0; hst[i] = 0.0f; }
  __syncthreads();

  for (int t = 0; t < TT; ++t) {
    bf16x16 ax[4], ah[4];
    const unsigned short* xbase = gseq + ((size_t)t * BB + b0) * HH;
    for (int kt = 0; kt < 4; ++kt) {
      const unsigned short* r = xbase + (size_t)mrow * HH + kt * 32 + khalf;
      TileBF tx;
      tx.q[0] = *(const uint4*)(r);
      tx.q[1] = *(const uint4*)(r + 16);
      ax[kt] = tx.v;
      ah[kt] = load_tile_swz(hbf + kt * 512, lane);
    }
    // 24 tiles: gate = j>>3 (0:u, 1:r, 2:hc), nt = j&7; 3 per wave
    for (int s = 0; s < 3; ++s) {
      int j = wave * 3 + s;
      int gate = j >> 3, nt = j & 7;
      f32x8 acc = {};
      for (int kt = 0; kt < 4; ++kt) {
        int toff = (nt * 4 + kt) * 512;
        if (gate == 0) {
          acc = wmma_bf16(ax[kt], load_tile_swz(wu + toff, lane), acc);
          acc = wmma_bf16(ah[kt], load_tile_swz(uu + toff, lane), acc);
        } else if (gate == 1) {
          acc = wmma_bf16(ax[kt], load_tile_swz(wr + toff, lane), acc);
          acc = wmma_bf16(ah[kt], load_tile_swz(ur + toff, lane), acc);
        } else {
          acc = wmma_bf16(ah[kt], load_tile_swz(uc + toff, lane), acc);
        }
      }
      int col = gate * HH + nt * 16 + (lane & 15);
      for (int i = 0; i < 8; ++i) gates[(rbase + i) * G3 + col] = acc[i];
    }
    __syncthreads();
    for (int e = tid; e < 16 * HH; e += 256) {
      int m = e >> 7, h = e & 127;
      float u = sigmoidf(gates[m * G3 + h] + bu[h]);
      float r = sigmoidf(gates[m * G3 + HH + h] + br[h]);
      float c = tanhf(r * gates[m * G3 + 2*HH + h]);
      float a = attn[(size_t)(b0 + m) * TT + t];
      float u_ = a * u;
      float hn = (1.0f - u_) * hst[e] + u_ * c;
      hst[e] = hn;
      hbf[swz_index(h, m, HH)] = f32_to_bf16(hn);
    }
    __syncthreads();
  }
  for (int e = tid; e < 16 * HH; e += 256) {
    int m = e >> 7, h = e & 127;
    hist_hid[(size_t)(b0 + m) * HH + h] = hst[e];
  }
}

// ---------------------------------------------------------------------------
// K5: BN -> fc1(sig) -> fc2(sig) -> fc3 + item_bias; outputs (logit, sigmoid)
// ---------------------------------------------------------------------------
__global__ __launch_bounds__(128) void mlp_kernel(
    const int* __restrict__ user, const int* __restrict__ item,
    const float* __restrict__ user_emb_table, const float* __restrict__ item_bias_table,
    const float* __restrict__ item_join, const float* __restrict__ hist_hid,
    const float* __restrict__ gamma, const float* __restrict__ beta,
    const float* __restrict__ mean, const float* __restrict__ var,
    const float* __restrict__ w1, const float* __restrict__ b1,
    const float* __restrict__ w2, const float* __restrict__ b2,
    const float* __restrict__ w3, const float* __restrict__ b3,
    float* __restrict__ out) {
  __shared__ float x[JOIN];
  __shared__ float h1[80];
  __shared__ float h2[40];
  int b = blockIdx.x, tid = threadIdx.x;
  int u = user[b];
  for (int j = tid; j < JOIN; j += 128) {
    float v;
    if (j < 64)       v = user_emb_table[(size_t)u * 64 + j];
    else if (j < 192) v = item_join[(size_t)b * HH + (j - 64)];
    else              v = hist_hid[(size_t)b * HH + (j - 192)];
    x[j] = (v - mean[j]) * rsqrtf(var[j] + 1e-3f) * gamma[j] + beta[j];
  }
  __syncthreads();
  if (tid < 80) {
    float a = b1[tid];
    for (int k = 0; k < JOIN; ++k) a += x[k] * w1[k * 80 + tid];
    h1[tid] = sigmoidf(a);
  }
  __syncthreads();
  if (tid < 40) {
    float a = b2[tid];
    for (int k = 0; k < 80; ++k) a += h1[k] * w2[k * 40 + tid];
    h2[tid] = sigmoidf(a);
  }
  __syncthreads();
  if (tid == 0) {
    float a = b3[0];
    for (int k = 0; k < 40; ++k) a += h2[k] * w3[k];
    a += item_bias_table[item[b]];
    out[b] = a;
    out[BB + b] = sigmoidf(a);
  }
}

// ---------------------------------------------------------------------------
// Host launcher
// ---------------------------------------------------------------------------
extern "C" void kernel_launch(void* const* d_in, const int* in_sizes, int n_in,
                              void* d_out, int out_size, void* d_ws, size_t ws_size,
                              hipStream_t stream) {
  (void)in_sizes; (void)n_in; (void)out_size; (void)ws_size;

  const int*   user      = (const int*)d_in[0];
  const int*   item      = (const int*)d_in[1];
  const int*   history   = (const int*)d_in[2];
  /* d_in[3] = length: unused by reference */
  const int*   cate_list = (const int*)d_in[4];
  const float* user_emb  = (const float*)d_in[5];
  const float* item_emb  = (const float*)d_in[6];
  const float* cate_emb  = (const float*)d_in[7];
  const float* item_bias = (const float*)d_in[8];
  const float* gru_wx    = (const float*)d_in[9];
  const float* gru_wh    = (const float*)d_in[10];
  const float* gru_b_in  = (const float*)d_in[11];
  const float* gru_b_rec = (const float*)d_in[12];
  const float* au_wu     = (const float*)d_in[13];
  const float* au_bu     = (const float*)d_in[14];
  const float* au_uu     = (const float*)d_in[15];
  const float* au_wr     = (const float*)d_in[16];
  const float* au_br     = (const float*)d_in[17];
  const float* au_ur     = (const float*)d_in[18];
  const float* au_uc     = (const float*)d_in[19];
  const float* bn_gamma  = (const float*)d_in[20];
  const float* bn_beta   = (const float*)d_in[21];
  const float* bn_mean   = (const float*)d_in[22];
  const float* bn_var    = (const float*)d_in[23];
  const float* fc1_w     = (const float*)d_in[24];
  const float* fc1_b     = (const float*)d_in[25];
  const float* fc2_w     = (const float*)d_in[26];
  const float* fc2_b     = (const float*)d_in[27];
  const float* fc3_w     = (const float*)d_in[28];
  const float* fc3_b     = (const float*)d_in[29];

  // ---- workspace carving (256B aligned) ----
  unsigned char* ws = (unsigned char*)d_ws;
  size_t off = 0;
  auto carve = [&](size_t bytes) -> unsigned char* {
    unsigned char* p = ws + off;
    off = (off + bytes + 255) & ~(size_t)255;
    return p;
  };
  unsigned short* histj   = (unsigned short*)carve((size_t)BB * TT * HH * 2);  // 100 MB
  unsigned short* gseq    = (unsigned short*)carve((size_t)TT * BB * HH * 2);  // 100 MB
  float*          itjoin  = (float*)carve((size_t)BB * HH * 4);
  float*          attn    = (float*)carve((size_t)BB * TT * 4);
  float*          hhid    = (float*)carve((size_t)BB * HH * 4);
  unsigned short* wx_bf   = (unsigned short*)carve((size_t)HH * G3 * 2);
  unsigned short* wh_bf   = (unsigned short*)carve((size_t)HH * G3 * 2);
  unsigned short* wu_bf   = (unsigned short*)carve((size_t)HH * HH * 2);
  unsigned short* uu_bf   = (unsigned short*)carve((size_t)HH * HH * 2);
  unsigned short* wr_bf   = (unsigned short*)carve((size_t)HH * HH * 2);
  unsigned short* ur_bf   = (unsigned short*)carve((size_t)HH * HH * 2);
  unsigned short* uc_bf   = (unsigned short*)carve((size_t)HH * HH * 2);

  // ---- K0: weight conversion ----
  const int nW3 = HH * G3, nW1 = HH * HH;
  cvt_bf16_kernel<<<(nW3 + 255) / 256, 256, 0, stream>>>(gru_wx, wx_bf, nW3);
  cvt_bf16_kernel<<<(nW3 + 255) / 256, 256, 0, stream>>>(gru_wh, wh_bf, nW3);
  cvt_bf16_kernel<<<(nW1 + 255) / 256, 256, 0, stream>>>(au_wu, wu_bf, nW1);
  cvt_bf16_kernel<<<(nW1 + 255) / 256, 256, 0, stream>>>(au_uu, uu_bf, nW1);
  cvt_bf16_kernel<<<(nW1 + 255) / 256, 256, 0, stream>>>(au_wr, wr_bf, nW1);
  cvt_bf16_kernel<<<(nW1 + 255) / 256, 256, 0, stream>>>(au_ur, ur_bf, nW1);
  cvt_bf16_kernel<<<(nW1 + 255) / 256, 256, 0, stream>>>(au_uc, uc_bf, nW1);

  // ---- K1: gathers ----
  gather_hist_kernel<<<BB * TT, 128, 0, stream>>>(history, cate_list, item_emb, cate_emb, histj);
  gather_item_kernel<<<BB, 128, 0, stream>>>(item, cate_list, item_emb, cate_emb, itjoin);

  // ---- K2: fused GRU (348 KB dynamic LDS requested per block, 252 KB used; CDNA5 WGP has 320 KB) ----
  const size_t gruLds = (size_t)(HH * G3 * 2 + HH * G3 + 16 * HH) * 2   // shorts
                      + (size_t)(16 * HH + 2 * 16 * G3) * 4;            // floats
  hipFuncSetAttribute((const void*)gru_kernel,
                      hipFuncAttributeMaxDynamicSharedMemorySize, (int)gruLds);
  gru_kernel<<<BB / 16, 256, gruLds, stream>>>(histj, wx_bf, wh_bf, gru_b_in, gru_b_rec, gseq);

  // ---- K3: attention softmax ----
  attn_kernel<<<BB, 256, 0, stream>>>(gseq, itjoin, attn);

  // ---- K4: fused AUGRU (196 KB dynamic LDS) ----
  const size_t auLds = (size_t)(5 * HH * HH + 16 * HH) * 2
                     + (size_t)(16 * HH + 16 * G3) * 4;
  hipFuncSetAttribute((const void*)augru_kernel,
                      hipFuncAttributeMaxDynamicSharedMemorySize, (int)auLds);
  augru_kernel<<<BB / 16, 256, auLds, stream>>>(gseq, attn, wu_bf, uu_bf, wr_bf, ur_bf, uc_bf,
                                                au_bu, au_br, hhid);

  // ---- K5: final MLP ----
  mlp_kernel<<<BB, 128, 0, stream>>>(user, item, user_emb, item_bias, itjoin, hhid,
                                     bn_gamma, bn_beta, bn_mean, bn_var,
                                     fc1_w, fc1_b, fc2_w, fc2_b, fc3_w, fc3_b,
                                     (float*)d_out);
}
